// GraphAttentionLayer_58402965291610
// MI455X (gfx1250) — compile-verified, matrix operand order checked by reading, not run on hardware
//
#include <hip/hip_runtime.h>

// ---- CDNA5 (gfx1250, wave32) vector types ----
typedef __attribute__((ext_vector_type(16))) __bf16 v16bf;
typedef __attribute__((ext_vector_type(8)))  __bf16 v8bf;
typedef __attribute__((ext_vector_type(2)))  __bf16 v2bf;
typedef __attribute__((ext_vector_type(8)))  float  v8f;
typedef __attribute__((ext_vector_type(8)))  float  f32x8;

#define NN    8192
#define FF    256
#define L2E   1.44269504088896340736f
#define NEGI  (-9.0e15f)

// =====================================================================
// Kernel 1: h = x @ W^T  via v_wmma_f32_16x16x32_bf16, store h TRANSPOSED
// (hT[n][i], bf16) so the attention matmul's B-fragments are contiguous.
// One 16x16 tile per wave; K = 256 -> 8 WMMA steps.
//   A tile (16x32): lane holds row M=lane%16, K-octets (lane/16)*8, 16+(lane/16)*8
//   B tile (32x16): B[k][n] = W[n][k] -> lane holds W row n = n0+lane%16, same octets
// =====================================================================
__global__ __launch_bounds__(256) void gat_h_gemm(
    const float* __restrict__ x, const float* __restrict__ W,
    __bf16* __restrict__ hT)
{
    const int lane = threadIdx.x & 31;
    const int wave = threadIdx.x >> 5;
    const int lr   = lane & 15;
    const int half = lane >> 4;

    const int tile = blockIdx.x * 8 + wave;      // 512 * 16 = 8192 tiles
    const int i0 = (tile >> 4) * 16;
    const int n0 = (tile & 15) * 16;

    const float* __restrict__ xrow = x + (size_t)(i0 + lr) * FF;
    const float* __restrict__ wrow = W + (size_t)(n0 + lr) * FF;

    v8f c = {};
#pragma unroll
    for (int kk = 0; kk < 8; ++kk) {
        const int k = kk * 32 + half * 8;
        f32x8 ax0 = *(const f32x8*)(xrow + k);        // K = k..k+7
        f32x8 ax1 = *(const f32x8*)(xrow + k + 16);   // K = k+16..k+23
        f32x8 bw0 = *(const f32x8*)(wrow + k);
        f32x8 bw1 = *(const f32x8*)(wrow + k + 16);
        v16bf af, bfrg;
#pragma unroll
        for (int e = 0; e < 8; ++e) {
            af[e]       = (__bf16)ax0[e];
            af[8 + e]   = (__bf16)ax1[e];
            bfrg[e]     = (__bf16)bw0[e];
            bfrg[8 + e] = (__bf16)bw1[e];
        }
        c = __builtin_amdgcn_wmma_f32_16x16x32_bf16(false, af, false, bfrg,
                                                    (short)0, c, false, false);
    }
    // C layout: N = lane%16, M = v + 8*(lane/16). Store transposed: hT[n][i].
#pragma unroll
    for (int v = 0; v < 8; ++v)
        hT[(size_t)(n0 + lr) * NN + i0 + v + half * 8] = (__bf16)c[v];
}

// =====================================================================
// Kernel 2: s1[j] = h[j]·a1, s2[j] = h[j]·a2  (from hT, coalesced over j)
// =====================================================================
__global__ __launch_bounds__(256) void gat_scores(
    const __bf16* __restrict__ hT, const float* __restrict__ a,
    float* __restrict__ s1, float* __restrict__ s2)
{
    const int j = blockIdx.x * 256 + threadIdx.x;
    float acc1 = 0.0f, acc2 = 0.0f;
#pragma unroll 4
    for (int n = 0; n < FF; ++n) {
        float hv = (float)hT[(size_t)n * NN + j];
        acc1 = fmaf(hv, a[n], acc1);
        acc2 = fmaf(hv, a[FF + n], acc2);
    }
    s1[j] = acc1;
    s2[j] = acc2;
}

// =====================================================================
// Kernel 3: one wave per row: stream adj row once (the only 256MB HBM pass),
// ballot-pack it to a 1-bit mask, exact online softmax stats (m_i, 1/l_i).
// The running-max rescale exp lives in a rarely-taken (EXEC-skipped) branch
// so the steady state issues ~1 v_exp_f32 per element.
// =====================================================================
__global__ __launch_bounds__(256) void gat_softmax_stats(
    const int* __restrict__ adj, const float* __restrict__ s1,
    const float* __restrict__ s2, float* __restrict__ mrow,
    float* __restrict__ invl, unsigned* __restrict__ mask)
{
    const int lane = threadIdx.x & 31;
    const int wave = threadIdx.x >> 5;
    const int i = blockIdx.x * 8 + wave;

    const float sival = s1[i];
    const int* __restrict__ arow = adj + (size_t)i * NN;
    unsigned* __restrict__ mrp = mask + (size_t)i * (NN / 32);

    float m = NEGI, l = 0.0f;
    for (int jb = 0; jb < NN / 32; ++jb) {
        const int j = jb * 32 + lane;
        __builtin_prefetch(arow + j + 4096, 0, 0);   // global_prefetch_b8 ahead of stream
        const int av = arow[j];
        float ev = sival + s2[j];
        ev = fmaxf(ev, 0.2f * ev);                   // leaky_relu, branchless
        const bool on = av > 0;
        const unsigned bal = (unsigned)__ballot(on);
        if (lane == 0) mrp[jb] = bal;
        const float eeff = on ? ev : NEGI;
        if (eeff > m) {                              // rare after warmup -> execz skip
            l *= __builtin_amdgcn_exp2f((m - eeff) * L2E);
            m = eeff;
        }
        l += on ? __builtin_amdgcn_exp2f((eeff - m) * L2E) : 0.0f;
    }
    // wave32 butterfly combine of (m, l)
#pragma unroll
    for (int off = 16; off >= 1; off >>= 1) {
        const float mo = __shfl_xor(m, off, 32);
        const float lo = __shfl_xor(l, off, 32);
        const float mn = fmaxf(m, mo);
        l = l * __builtin_amdgcn_exp2f((m - mn) * L2E)
          + lo * __builtin_amdgcn_exp2f((mo - mn) * L2E);
        m = mn;
    }
    if (lane == 0) {
        mrow[i] = m;
        invl[i] = (l > 0.0f) ? __builtin_amdgcn_rcpf(l) : 0.0f;
    }
}

// =====================================================================
// Kernel 4: fused h' = softmax(att) @ h. 16 rows per workgroup, 8 waves x
// 32 output columns. The 16x32 P tile is computed ONCE per workgroup
// (2 p-values per thread), staged in double-buffered LDS (2KB), and each
// wave pulls its A-fragment with two ds_load_b128 -> 8x less exp/cvt work
// than the replicated version. One barrier per 32-j block.
// =====================================================================
__global__ __launch_bounds__(256) void gat_attn_matmul(
    const float* __restrict__ s1, const float* __restrict__ s2,
    const float* __restrict__ mrow, const float* __restrict__ invl,
    const unsigned* __restrict__ mask, const __bf16* __restrict__ hT,
    float* __restrict__ out)
{
    __shared__ alignas(16) __bf16 pbuf[2][16][32];   // [buf][row][j] : 2 KB

    const int tid  = threadIdx.x;
    const int lane = tid & 31;
    const int wave = tid >> 5;
    const int lr   = lane & 15;
    const int half = lane >> 4;

    const int i0 = blockIdx.x * 16;
    const int n0 = wave * 32;

    // ---- producer role: thread t computes P[pr][pj], P[pr][pj+1] ----
    const int pr = tid >> 4;            // 0..15  (row in tile)
    const int pj = (tid & 15) * 2;      // 0,2,..,30 (j pair in block)
    const float psr = s1[i0 + pr];
    const float pmr = mrow[i0 + pr];
    const float pir = invl[i0 + pr];
    const unsigned* __restrict__ pmask = mask + (size_t)(i0 + pr) * (NN / 32);

    // ---- consumer role: B-fragment row pointers (L2-resident hT) ----
    const __bf16* __restrict__ b0row = hT + (size_t)(n0 + lr) * NN;
    const __bf16* __restrict__ b1row = hT + (size_t)(n0 + 16 + lr) * NN;

    v8f c0 = {}, c1 = {};

    auto produce = [&](int jb) {
        const unsigned mw = pmask[jb];
        float e0 = psr + s2[jb * 32 + pj];
        float e1 = psr + s2[jb * 32 + pj + 1];
        e0 = fmaxf(e0, 0.2f * e0);
        e1 = fmaxf(e1, 0.2f * e1);
        const float p0 = ((mw >> pj) & 1u)
            ? __builtin_amdgcn_exp2f((e0 - pmr) * L2E) * pir : 0.0f;
        const float p1 = ((mw >> (pj + 1)) & 1u)
            ? __builtin_amdgcn_exp2f((e1 - pmr) * L2E) * pir : 0.0f;
        v2bf pk; pk[0] = (__bf16)p0; pk[1] = (__bf16)p1;   // v_cvt_pk_bf16_f32
        *(v2bf*)(&pbuf[jb & 1][pr][pj]) = pk;              // single ds_store_b32
    };

    produce(0);
    __syncthreads();

    for (int jb = 0; jb < NN / 32; ++jb) {
        // A fragment from LDS: row lr, octets half*8 and 16+half*8 (ds_load_b128 x2)
        const __bf16* prow = &pbuf[jb & 1][lr][0];
        v8bf pa = *(const v8bf*)(prow + half * 8);
        v8bf pb = *(const v8bf*)(prow + 16 + half * 8);
        v16bf afrag = __builtin_shufflevector(pa, pb,
                        0,1,2,3,4,5,6,7,8,9,10,11,12,13,14,15);

        // B fragments: contiguous 16B chunks from hT rows
        const int jbase = jb * 32 + half * 8;
        v8bf q0a = *(const v8bf*)(b0row + jbase);
        v8bf q0b = *(const v8bf*)(b0row + jbase + 16);
        v8bf q1a = *(const v8bf*)(b1row + jbase);
        v8bf q1b = *(const v8bf*)(b1row + jbase + 16);
        v16bf bf0 = __builtin_shufflevector(q0a, q0b,
                        0,1,2,3,4,5,6,7,8,9,10,11,12,13,14,15);
        v16bf bf1 = __builtin_shufflevector(q1a, q1b,
                        0,1,2,3,4,5,6,7,8,9,10,11,12,13,14,15);

        // overlap next tile's exp/cvt with this tile's WMMA + loads
        if (jb + 1 < NN / 32) produce(jb + 1);

        c0 = __builtin_amdgcn_wmma_f32_16x16x32_bf16(false, afrag, false, bf0,
                                                     (short)0, c0, false, false);
        c1 = __builtin_amdgcn_wmma_f32_16x16x32_bf16(false, afrag, false, bf1,
                                                     (short)0, c1, false, false);
        __syncthreads();
    }

    // C layout: N = lane%16, M = v + 8*(lane/16)
#pragma unroll
    for (int v = 0; v < 8; ++v) {
        const int row = i0 + v + half * 8;
        out[(size_t)row * FF + n0 + lr]      = c0[v];
        out[(size_t)row * FF + n0 + 16 + lr] = c1[v];
    }
}

// =====================================================================
extern "C" void kernel_launch(void* const* d_in, const int* in_sizes, int n_in,
                              void* d_out, int out_size, void* d_ws, size_t ws_size,
                              hipStream_t stream) {
    const float* x   = (const float*)d_in[0];   // 8192 x 256
    const int*   adj = (const int*)  d_in[1];   // 8192 x 8192
    const float* W   = (const float*)d_in[2];   // 256 x 256
    const float* a   = (const float*)d_in[3];   // 512
    float* out = (float*)d_out;                 // 8192 x 256

    // workspace layout: hT (4MB bf16) | s1,s2,m,inv (4x32KB) | mask (8MB)
    __bf16* hT  = (__bf16*)d_ws;
    float* s1   = (float*)((char*)d_ws + (size_t)FF * NN * sizeof(__bf16));
    float* s2   = s1 + NN;
    float* mrow = s2 + NN;
    float* invl = mrow + NN;
    unsigned* mask = (unsigned*)(invl + NN);

    gat_h_gemm       <<<1024, 256, 0, stream>>>(x, W, hT);
    gat_scores       <<<NN / 256, 256, 0, stream>>>(hT, a, s1, s2);
    gat_softmax_stats<<<1024, 256, 0, stream>>>(adj, s1, s2, mrow, invl, mask);
    gat_attn_matmul  <<<NN / 16, 256, 0, stream>>>(s1, s2, mrow, invl, mask, hT, out);
}
